// RBM_83408264888340
// MI455X (gfx1250) — compile-verified, matrix operand order checked by reading, not run on hardware
//
#include <hip/hip_runtime.h>
#include <hip/hip_bf16.h>
#include <math.h>

// ---------------------------------------------------------------------------
// RBM on gfx1250 (MI455X), wave32 / WMMA:
//   K1: fused bf16-WMMA GEMM (8192x2048x2048) + bias + zeta/sigmoid epilogue
//   K2: sequential Gibbs chain (2010 steps), one 1024-thread workgroup
//   K3: fused bf16-WMMA GEMM (8192x1024x1024): sum((zeta_v@W) * zeta_h)
//   K4: deterministic tree reductions + 10-sample free energy -> kl
//
// GEMM: block tile 64(M)x128(N), BK=32, 256 threads = 8 waves (2x4), wave tile
// 32x32 = 4 WMMA accs. A and B staged K-major in LDS (B transposed on write):
// every fragment is exactly 2x ds_load_b128. Tiles are streamed with
// GLOBAL_LOAD_ASYNC_TO_LDS_B128 into double-buffered fp32 LDS staging
// (ASYNCcnt-tracked, overlapped with WMMA); falls back to a register-staged
// software pipeline if the async builtins are absent.
// ---------------------------------------------------------------------------

typedef __attribute__((ext_vector_type(16))) __bf16 v16bf;
typedef __attribute__((ext_vector_type(8)))  __bf16 v8bf;
typedef __attribute__((ext_vector_type(4)))  __bf16 v4bf;
typedef __attribute__((ext_vector_type(8)))  float  v8f;
typedef int v4i_async __attribute__((vector_size(16)));   // builtin's pointee type

#define VIS    1024
#define HID    1024
#define DIMX   2048
#define NROWS  8192
#define KDIM   2048
#define TEMP_INV 5.0f        // 1 / 0.2
#define TOTAL_STEPS 2010
#define KL_SAMPLES 10

#define BM 64
#define BN 128
#define BK 32
#define LPAD 40              // bf16 row pitch: 80B, 16B-aligned, conflict-free

// workspace layout (floats)
#define WS_P1   0            // 2048 partials from K1  (16*128 blocks)
#define WS_P2   2048         // 1024 partials from K3  ( 8*128 blocks)
#define WS_SAMP 3072         // 10 * 2048 gibbs samples [sv(1024) | sh(1024)]

#if defined(__has_builtin)
#if __has_builtin(__builtin_amdgcn_global_load_async_to_lds_b128) && \
    __has_builtin(__builtin_amdgcn_s_wait_asynccnt)
#define HAVE_ASYNC_LDS 1
#else
#define HAVE_ASYNC_LDS 0
#endif
#else
#define HAVE_ASYNC_LDS 0
#endif

// ---------------------------------------------------------------------------
// helpers
// ---------------------------------------------------------------------------
__device__ __forceinline__ v16bf frag_rowmajor(const __bf16* rowbase, int hlf)
{
    // ISA 7.12.2 16-bit A/B frag: e=0..7 -> k=hlf*8+e ; e=8..15 -> k=16+hlf*8+(e-8)
    v8bf lo = *(const v8bf*)(rowbase + hlf * 8);
    v8bf hi = *(const v8bf*)(rowbase + 16 + hlf * 8);
    return __builtin_shufflevector(lo, hi, 0, 1, 2, 3, 4, 5, 6, 7,
                                           8, 9, 10, 11, 12, 13, 14, 15);
}

#define WMMA_BF16(a, b, c) \
    __builtin_amdgcn_wmma_f32_16x16x32_bf16(false, (a), false, (b), (short)0, (c), false, false)

// Pack A row chunk (8 fp32 -> v8bf, one ds_store_b128)
__device__ __forceinline__ void pack_a8(float4 f0, float4 f1, __bf16* dst)
{
    v8bf o;
    o[0] = (__bf16)f0.x; o[1] = (__bf16)f0.y; o[2] = (__bf16)f0.z; o[3] = (__bf16)f0.w;
    o[4] = (__bf16)f1.x; o[5] = (__bf16)f1.y; o[6] = (__bf16)f1.z; o[7] = (__bf16)f1.w;
    *(v8bf*)dst = o;
}

// 4x4 micro-transpose of B into Bt[n][k] (four ds_store_b64)
__device__ __forceinline__ void pack_bt4x4(const float4 f[4], int nb, int kb,
                                           __bf16 (*Bt)[LPAD])
{
    v4bf c0, c1, c2, c3;
    c0[0] = (__bf16)f[0].x; c0[1] = (__bf16)f[1].x; c0[2] = (__bf16)f[2].x; c0[3] = (__bf16)f[3].x;
    c1[0] = (__bf16)f[0].y; c1[1] = (__bf16)f[1].y; c1[2] = (__bf16)f[2].y; c1[3] = (__bf16)f[3].y;
    c2[0] = (__bf16)f[0].z; c2[1] = (__bf16)f[1].z; c2[2] = (__bf16)f[2].z; c2[3] = (__bf16)f[3].z;
    c3[0] = (__bf16)f[0].w; c3[1] = (__bf16)f[1].w; c3[2] = (__bf16)f[2].w; c3[3] = (__bf16)f[3].w;
    *(v4bf*)&Bt[nb + 0][kb] = c0;
    *(v4bf*)&Bt[nb + 1][kb] = c1;
    *(v4bf*)&Bt[nb + 2][kb] = c2;
    *(v4bf*)&Bt[nb + 3][kb] = c3;
}

#if HAVE_ASYNC_LDS
// one 16B async copy: global -> LDS (ASYNCcnt-tracked). Builtin signature is
// (v4i AS1*, v4i AS3*, imm offset, imm cpol). Generic LDS addresses truncate
// to the 32-bit LDS offset per the ISA aperture rule.
__device__ __forceinline__ void async_cp16(const void* g, void* l)
{
    __builtin_amdgcn_global_load_async_to_lds_b128(
        (__attribute__((address_space(1))) v4i_async*)(unsigned long long)g,
        (__attribute__((address_space(3))) v4i_async*)(unsigned int)(unsigned long long)l,
        0, 0);
}
#endif

// ---------------------------------------------------------------------------
// Software-pipelined bf16 WMMA block-GEMM core (wave tile 32x32 -> acc[2][2]).
// ---------------------------------------------------------------------------
__device__ __forceinline__ void gemm_core(const float* __restrict__ Ag,
                                          const float* __restrict__ Bg,
                                          int m0, int n0, int kdim,
                                          int apitch, int bpitch,
                                          v8f acc[2][2])
{
    __shared__ __attribute__((aligned(16))) __bf16 As[BM][LPAD];
    __shared__ __attribute__((aligned(16))) __bf16 Bt[BN][LPAD];

    const int tid  = threadIdx.x;
    const int lane = tid & 31;
    const int wave = tid >> 5;
    const int wmw  = wave >> 2;          // 0..1
    const int wnw  = wave & 3;           // 0..3
    const int hlf  = lane >> 4;
    const int l16  = lane & 15;

    // per-thread staging geometry
    const int ar = tid >> 2, akc = (tid & 3) << 3;         // A: row, 8-wide k chunk
    const int kb = (tid >> 5) << 2, nb = (tid & 31) << 2;  // B: 4x4 micro-tile

#if HAVE_ASYNC_LDS
    __shared__ __attribute__((aligned(16))) float Afs[2][BM * BK];
    __shared__ __attribute__((aligned(16))) float Bfs[2][BK * BN];

    // issue async copies of one K-chunk into staging buffer `buf`
    auto issue = [&](int k0, int buf) {
#pragma unroll
        for (int p = 0; p < 2; ++p) {                // A: 64x32 fp32 = 512 x 16B
            int e = tid + p * 256;
            int r = e >> 3, kc = (e & 7) << 2;
            async_cp16(&Ag[(size_t)(m0 + r) * apitch + k0 + kc],
                       &Afs[buf][r * BK + kc]);
        }
#pragma unroll
        for (int p = 0; p < 4; ++p) {                // B: 32x128 fp32 = 1024 x 16B
            int e = tid + p * 256;
            int kr = e >> 5, nc = (e & 31) << 2;
            async_cp16(&Bg[(size_t)(k0 + kr) * bpitch + n0 + nc],
                       &Bfs[buf][kr * BN + nc]);
        }
    };

    issue(0, 0);
    __builtin_amdgcn_s_wait_asynccnt(0);
    __syncthreads();

    for (int k0 = 0; k0 < kdim; k0 += BK) {
        const int cur = (k0 >> 5) & 1;
        const bool more = (k0 + BK) < kdim;
        if (more) issue(k0 + BK, cur ^ 1);           // DMA next chunk, overlapped

        // convert fp32 staging -> bf16 K-major tiles (incl. B transpose)
        {
            const float* Af = Afs[cur];
            float4 f0 = *(const float4*)&Af[ar * BK + akc];
            float4 f1 = *(const float4*)&Af[ar * BK + akc + 4];
            pack_a8(f0, f1, &As[ar][akc]);
            const float* Bf = Bfs[cur];
            float4 fb[4];
#pragma unroll
            for (int j = 0; j < 4; ++j)
                fb[j] = *(const float4*)&Bf[(kb + j) * BN + nb];
            pack_bt4x4(fb, nb, kb, Bt);
        }
        __syncthreads();

        v16bf a0 = frag_rowmajor(&As[wmw * 32 + l16][0], hlf);
        v16bf a1 = frag_rowmajor(&As[wmw * 32 + 16 + l16][0], hlf);
        v16bf b0 = frag_rowmajor(&Bt[wnw * 32 + l16][0], hlf);
        v16bf b1 = frag_rowmajor(&Bt[wnw * 32 + 16 + l16][0], hlf);
        acc[0][0] = WMMA_BF16(a0, b0, acc[0][0]);
        acc[0][1] = WMMA_BF16(a0, b1, acc[0][1]);
        acc[1][0] = WMMA_BF16(a1, b0, acc[1][0]);
        acc[1][1] = WMMA_BF16(a1, b1, acc[1][1]);

        if (more) __builtin_amdgcn_s_wait_asynccnt(0);
        __syncthreads();
    }
#else
    // fallback: register-staged pipeline (global float4 loads one chunk ahead)
    float4 aR0, aR1, bR[4];
    auto loadrg = [&](int k0) {
        aR0 = *(const float4*)&Ag[(size_t)(m0 + ar) * apitch + k0 + akc];
        aR1 = *(const float4*)&Ag[(size_t)(m0 + ar) * apitch + k0 + akc + 4];
#pragma unroll
        for (int j = 0; j < 4; ++j)
            bR[j] = *(const float4*)&Bg[(size_t)(k0 + kb + j) * bpitch + n0 + nb];
    };

    loadrg(0);
    for (int k0 = 0; k0 < kdim; k0 += BK) {
        pack_a8(aR0, aR1, &As[ar][akc]);
        pack_bt4x4(bR, nb, kb, Bt);
        __syncthreads();

        if (k0 + BK < kdim) loadrg(k0 + BK);

        v16bf a0 = frag_rowmajor(&As[wmw * 32 + l16][0], hlf);
        v16bf a1 = frag_rowmajor(&As[wmw * 32 + 16 + l16][0], hlf);
        v16bf b0 = frag_rowmajor(&Bt[wnw * 32 + l16][0], hlf);
        v16bf b1 = frag_rowmajor(&Bt[wnw * 32 + 16 + l16][0], hlf);
        acc[0][0] = WMMA_BF16(a0, b0, acc[0][0]);
        acc[0][1] = WMMA_BF16(a0, b1, acc[0][1]);
        acc[1][0] = WMMA_BF16(a1, b0, acc[1][0]);
        acc[1][1] = WMMA_BF16(a1, b1, acc[1][1]);
        __syncthreads();
    }
#endif
}

__device__ __forceinline__ float block_reduce_256(float v, float* red)
{
    const int t = threadIdx.x;
    red[t] = v; __syncthreads();
    for (int off = 128; off > 0; off >>= 1) {
        if (t < off) red[t] += red[t + off];
        __syncthreads();
    }
    float r = red[0];
    __syncthreads();
    return r;
}

// ---------------------------------------------------------------------------
// K1: logits GEMM + zeta/log-posterior epilogue
// ---------------------------------------------------------------------------
__global__ __launch_bounds__(256)
void rbm_k1_logits_zeta(const float* __restrict__ vecs,
                        const float* __restrict__ lw,
                        const float* __restrict__ lb,
                        const float* __restrict__ bv,
                        const float* __restrict__ bh,
                        const float* __restrict__ eps,
                        float* __restrict__ zv_out,
                        float* __restrict__ zh_out,
                        float* __restrict__ partial1)
{
    __shared__ float red[256];
    const int tid  = threadIdx.x;
    const int lane = tid & 31;
    const int wave = tid >> 5;
    const int wmw  = wave >> 2;
    const int wnw  = wave & 3;
    const int hlf  = lane >> 4;
    const int l16  = lane & 15;
    const int m0   = blockIdx.y * BM;
    const int n0   = blockIdx.x * BN;

    v8f acc[2][2] = {};
    gemm_core(vecs, lw, m0, n0, KDIM, KDIM, DIMX, acc);

    float local = 0.0f;
#pragma unroll
    for (int sm = 0; sm < 2; ++sm) {
#pragma unroll
        for (int sn = 0; sn < 2; ++sn) {
#pragma unroll
            for (int v = 0; v < 8; ++v) {
                int r = m0 + wmw * 32 + sm * 16 + hlf * 8 + v;
                int c = n0 + wnw * 32 + sn * 16 + l16;
                float logit = acc[sm][sn][v] + lb[c];
                float z  = 1.0f / (1.0f + expf(-(logit + eps[(size_t)r * DIMX + c]) * TEMP_INV));
                float pr = 1.0f / (1.0f + expf(-logit));
                float bterm = (c < VIS) ? bv[c] : bh[c - VIS];
                local += logf(z * pr + (1.0f - z) * (1.0f - pr)) + z * bterm;
                if (c < VIS) zv_out[(size_t)r * VIS + c] = z;
                else         zh_out[(size_t)r * HID + (c - VIS)] = z;
            }
        }
    }

    float tot = block_reduce_256(local, red);
    if (tid == 0) partial1[blockIdx.y * gridDim.x + blockIdx.x] = tot;
}

// ---------------------------------------------------------------------------
// K2: sequential Gibbs chain (latency-bound; 32 waves on one WGP, state in LDS)
// ---------------------------------------------------------------------------
__global__ __launch_bounds__(1024)
void rbm_k2_gibbs(const float* __restrict__ w,
                  const float* __restrict__ bv,
                  const float* __restrict__ bh,
                  const float* __restrict__ u_init,
                  const float* __restrict__ u_h,
                  const float* __restrict__ u_v,
                  float* __restrict__ samples)
{
    __shared__ float sv[VIS];
    __shared__ float sh[HID];
    const int t = threadIdx.x;

    sv[t] = (u_init[t] >= 0.5f) ? 1.0f : 0.0f;
    __syncthreads();

    for (int step = 0; step < TOTAL_STEPS; ++step) {
        // h_j = bern(u, sigmoid(bh_j + sum_i sv_i * w[i][j]))  (coalesced over j)
        float s = bh[t];
#pragma unroll 8
        for (int i = 0; i < VIS; ++i) {
            if ((i & 63) == 0 && i + 64 < VIS)
                __builtin_prefetch(&w[(size_t)(i + 64) * HID + t], 0, 0);
            s = fmaf(sv[i], w[(size_t)i * HID + t], s);
        }
        float ph  = 1.0f / (1.0f + expf(-s));
        float shv = (u_h[(size_t)step * HID + t] >= ph) ? 1.0f : 0.0f;
        sh[t] = shv;
        __syncthreads();

        // v_i = bern(u, sigmoid(bv_i + sum_j sh_j * w[i][j]))  (row-contiguous)
        float s2 = bv[t];
        const float* wrow = w + (size_t)t * HID;
#pragma unroll 8
        for (int j = 0; j < HID; ++j) s2 = fmaf(sh[j], wrow[j], s2);
        float pv  = 1.0f / (1.0f + expf(-s2));
        float svv = (u_v[(size_t)step * VIS + t] >= pv) ? 1.0f : 0.0f;
        __syncthreads();
        sv[t] = svv;
        __syncthreads();

        if (step >= TOTAL_STEPS - KL_SAMPLES) {
            int sid = step - (TOTAL_STEPS - KL_SAMPLES);
            samples[(size_t)sid * DIMX + t]       = svv;
            samples[(size_t)sid * DIMX + VIS + t] = shv;
        }
    }
}

// ---------------------------------------------------------------------------
// K3: pairwise free-energy GEMM: sum over (r,n) of (zeta_v @ W)[r,n]*zeta_h[r,n]
// ---------------------------------------------------------------------------
__global__ __launch_bounds__(256)
void rbm_k3_fepair(const float* __restrict__ zv,
                   const float* __restrict__ w,
                   const float* __restrict__ zh,
                   float* __restrict__ partial2)
{
    __shared__ float red[256];
    const int tid  = threadIdx.x;
    const int lane = tid & 31;
    const int wave = tid >> 5;
    const int wmw  = wave >> 2;
    const int wnw  = wave & 3;
    const int hlf  = lane >> 4;
    const int l16  = lane & 15;
    const int m0   = blockIdx.y * BM;
    const int n0   = blockIdx.x * BN;

    v8f acc[2][2] = {};
    gemm_core(zv, w, m0, n0, VIS, VIS, HID, acc);

    float local = 0.0f;
#pragma unroll
    for (int sm = 0; sm < 2; ++sm) {
#pragma unroll
        for (int sn = 0; sn < 2; ++sn) {
#pragma unroll
            for (int v = 0; v < 8; ++v) {
                int r = m0 + wmw * 32 + sm * 16 + hlf * 8 + v;
                int c = n0 + wnw * 32 + sn * 16 + l16;
                local += acc[sm][sn][v] * zh[(size_t)r * HID + c];
            }
        }
    }

    float tot = block_reduce_256(local, red);
    if (tid == 0) partial2[blockIdx.y * gridDim.x + blockIdx.x] = tot;
}

// ---------------------------------------------------------------------------
// K4: deterministic finalize (fixed-order reductions -> bit-stable replays)
// ---------------------------------------------------------------------------
__global__ __launch_bounds__(256)
void rbm_k4_finalize(const float* __restrict__ w,
                     const float* __restrict__ bv,
                     const float* __restrict__ bh,
                     const float* __restrict__ ws,
                     float* __restrict__ kl_out)
{
    __shared__ float red[256];
    __shared__ float ssv[VIS];
    const int t = threadIdx.x;

    float s = 0.0f;
    for (int i = t; i < 2048; i += 256) s += ws[WS_P1 + i];
    for (int i = t; i < 1024; i += 256) s += ws[WS_P2 + i];
    float sumA = block_reduce_256(s, red);

    const float* samp = ws + WS_SAMP;
    float fs = 0.0f;
    for (int sid = 0; sid < KL_SAMPLES; ++sid) {
        const float* sv = samp + (size_t)sid * DIMX;
        const float* sh = sv + VIS;
        for (int i = t; i < VIS; i += 256) ssv[i] = sv[i];
        __syncthreads();
        for (int j = t; j < HID; j += 256) {
            float hj = 0.0f;
#pragma unroll 8
            for (int i = 0; i < VIS; ++i) hj = fmaf(ssv[i], w[(size_t)i * HID + j], hj);
            fs += sh[j] * (hj + bh[j]);
        }
        for (int i = t; i < VIS; i += 256) fs += ssv[i] * bv[i];
        __syncthreads();
    }
    float sumB = block_reduce_256(fs, red);

    if (t == 0)
        kl_out[0] = (sumA / (float)NROWS) - (sumB / (float)KL_SAMPLES);  // BETA=1
}

// ---------------------------------------------------------------------------
extern "C" void kernel_launch(void* const* d_in, const int* in_sizes, int n_in,
                              void* d_out, int out_size, void* d_ws, size_t ws_size,
                              hipStream_t stream)
{
    (void)in_sizes; (void)n_in; (void)out_size; (void)ws_size;

    const float* vecs     = (const float*)d_in[0];
    const float* logits_w = (const float*)d_in[1];
    const float* logits_b = (const float*)d_in[2];
    const float* w        = (const float*)d_in[3];
    const float* bv       = (const float*)d_in[4];
    const float* bh       = (const float*)d_in[5];
    const float* epsilon  = (const float*)d_in[6];
    const float* u_init   = (const float*)d_in[7];
    const float* u_h      = (const float*)d_in[8];
    const float* u_v      = (const float*)d_in[9];

    float* zv = (float*)d_out;                       // (8192, 1024)
    float* zh = zv + (size_t)NROWS * VIS;            // (8192, 1024)
    float* kl = zh + (size_t)NROWS * HID;            // scalar
    float* ws = (float*)d_ws;

    dim3 g1(DIMX / BN, NROWS / BM);                  // (16, 128)
    rbm_k1_logits_zeta<<<g1, 256, 0, stream>>>(vecs, logits_w, logits_b, bv, bh,
                                               epsilon, zv, zh, ws + WS_P1);

    rbm_k2_gibbs<<<1, 1024, 0, stream>>>(w, bv, bh, u_init, u_h, u_v, ws + WS_SAMP);

    dim3 g3(HID / BN, NROWS / BM);                   // (8, 128)
    rbm_k3_fepair<<<g3, 256, 0, stream>>>(zv, w, zh, ws + WS_P2);

    rbm_k4_finalize<<<1, 256, 0, stream>>>(w, bv, bh, ws, kl);
}
